// MagnitudeFlow_85031762526814
// MI455X (gfx1250) — matrix operand order repaired and checked by reading, not compiled, not run on hardware
//
#include <hip/hip_runtime.h>
#include <hip/hip_bf16.h>

// ---------------------------------------------------------------------------
// MagnitudeFlow fully-fused pipeline for MI455X (gfx1250, wave32, WMMA)
//   B=32768, DIM=768, NA=1024, K=8 patches, PD=16, RD=128, L=2, RA=16, TD=48, H=64
//   Single main kernel: ctx GEMMs + 2 relay layers + head, all WMMA-tiled at
//   16 rows/wave, with per-wave LDS marshalling between D- and A-fragment
//   layouts. Weights pre-packed into B-fragment lane layout (f16).
// ---------------------------------------------------------------------------

typedef _Float16 half_t;
typedef __attribute__((ext_vector_type(16))) _Float16 v16h;
typedef __attribute__((ext_vector_type(8)))  _Float16 v8h;
typedef __attribute__((ext_vector_type(8)))  float    v8f;

#define BATCH   32768
#define DIMV    768
#define NAV     1024
#define KP      8
#define PDV     16
#define RDV     128
#define TDV     48
#define HV      64

// workspace layout (bytes)
#define WS_PKEMB   0          // 24*4*512 halfs  =  98304 B
#define WS_PKTRI   98304      // 32*2*512 halfs  =  65536 B
#define WS_PKCTX   163840     //  4*8*512 halfs  =  32768 B
#define WS_PKW1    196608     // 16*8*512 halfs  = 131072 B
#define WS_PKW2    327680     // 16*2*512 halfs  =  32768 B
#define WS_ATTAB   360448     // 2*3*8*16*16 f32 =  49152 B
#define WS_PKAT    409600     // 16*3*512 halfs  =  49152 B

// dynamic LDS layout (bytes); per-wave strides noted
#define SM_CTX     0          // f16 [8][16][128]            = 32768
#define SM_XT      32768      // f32 [8][16][132]            = 67584
#define SM_LNT     100352     // f32 [8][16][132]            = 67584
#define SM_TRIT    167936     // f16 [8][16][72]             = 18432
#define SM_HHT     186368     // f16 [8][16][72]             = 18432
#define SM_PWT     204800     // f32 [8][16][17]             =  8704
#define SM_SN      213504     // f32 [8][16][8]              =  4096
#define SM_MC      217600     // f32 [128][9]                =  4608
#define SM_TOTAL   222208

__device__ inline float dev_gelu(float x) {
    return 0.5f * x * (1.0f + erff(x * 0.70710678118654752f));
}
__device__ inline float dev_sigmoid(float x) {
    return 1.0f / (1.0f + expf(-x));
}

// ---------------------------------------------------------------------------
// Prep 1: pack weights into WMMA B-fragment layout (f16) + slerp anchor table.
//   B-frag (16x16x32 f16): lane L supplies column N = L&15,
//   K = (e&7) + 8*(L>=16) + 16*(e>=8) for e in 0..15.
// ---------------------------------------------------------------------------
__global__ __launch_bounds__(256) void prep1_kernel(
    const float* __restrict__ embW,    // [768,64]
    const float* __restrict__ triW,    // [1024,32]
    const float* __restrict__ ctxW,    // [97,128]
    const float* __restrict__ w1,      // [2,8,48,64]
    const float* __restrict__ w2,      // [2,8,64,16]
    const float* __restrict__ home,    // [2,8,16,16]
    const float* __restrict__ anchors, // [2,8,16,16]
    half_t* __restrict__ pkEmb, half_t* __restrict__ pkTri,
    half_t* __restrict__ pkCtx, half_t* __restrict__ pkW1,
    half_t* __restrict__ pkW2, float* __restrict__ atTab)
{
    int i = blockIdx.x * 256 + threadIdx.x;
    if (i < 49152) {                       // emb_w: 24 k-steps x 4 n-tiles
        int fi = i >> 9, j = i & 511;
        int ks = fi >> 2, nt = fi & 3;
        int lane = j >> 4, e = j & 15;
        int k = ks * 32 + (e & 7) + ((lane >> 4) << 3) + ((e >> 3) << 4);
        int n = nt * 16 + (lane & 15);
        pkEmb[i] = (half_t)embW[k * 64 + n];
    } else if (i < 81920) {                // tri_w: 32 k-steps x 2 n-tiles
        int idx = i - 49152;
        int fi = idx >> 9, j = idx & 511;
        int ks = fi >> 1, nt = fi & 1;
        int lane = j >> 4, e = j & 15;
        int k = ks * 32 + (e & 7) + ((lane >> 4) << 3) + ((e >> 3) << 4);
        int n = nt * 16 + (lane & 15);
        pkTri[idx] = (half_t)triW[k * 32 + n];
    } else if (i < 98304) {                // ctx_w: K=97 zero-padded to 128; 4 x 8
        int idx = i - 81920;
        int fi = idx >> 9, j = idx & 511;
        int ks = fi >> 3, nt = fi & 7;
        int lane = j >> 4, e = j & 15;
        int k = ks * 32 + (e & 7) + ((lane >> 4) << 3) + ((e >> 3) << 4);
        int n = nt * 16 + (lane & 15);
        pkCtx[idx] = (half_t)((k < 97) ? ctxW[k * 128 + n] : 0.0f);
    } else if (i < 163840) {               // pw_w1: per (l,p): K=48 pad 64, N=64
        int idx = i - 98304;
        int frag = idx >> 9, j = idx & 511;
        int lp = frag >> 3, fi = frag & 7;
        int ks = fi >> 2, nt = fi & 3;
        int lane = j >> 4, e = j & 15;
        int k = ks * 32 + (e & 7) + ((lane >> 4) << 3) + ((e >> 3) << 4);
        int n = nt * 16 + (lane & 15);
        pkW1[idx] = (half_t)((k < TDV) ? w1[(lp * TDV + k) * HV + n] : 0.0f);
    } else if (i < 180224) {               // pw_w2: per (l,p): K=64, N=16
        int idx = i - 163840;
        int frag = idx >> 9, j = idx & 511;
        int lp = frag >> 1, ks = frag & 1;
        int lane = j >> 4, e = j & 15;
        int k = ks * 32 + (e & 7) + ((lane >> 4) << 3) + ((e >> 3) << 4);
        int n = lane & 15;
        pkW2[idx] = (half_t)w2[(lp * HV + k) * PDV + n];
    } else if (i < 180480) {               // slerp anchor table: 2*8*16 tasks
        int t = i - 180224;
        int l = t >> 7, rem = t & 127, p = rem >> 4, a = rem & 15;
        const float* hp = home    + ((((l * 8 + p) * 16) + a) << 4);
        const float* cp = anchors + ((((l * 8 + p) * 16) + a) << 4);
        float hv[16], cv[16], hs = 0.f, cs = 0.f;
        #pragma unroll
        for (int d = 0; d < 16; ++d) { hv[d] = hp[d]; cv[d] = cp[d]; hs += hv[d]*hv[d]; cs += cv[d]*cv[d]; }
        float hni = 1.0f / fmaxf(sqrtf(hs), 1e-12f);
        float cni = 1.0f / fmaxf(sqrtf(cs), 1e-12f);
        float dotc = 0.f;
        #pragma unroll
        for (int d = 0; d < 16; ++d) { hv[d] *= hni; cv[d] *= cni; dotc += hv[d]*cv[d]; }
        dotc = fminf(fmaxf(dotc, -1.0f + 1e-7f), 1.0f - 1e-7f);
        float omega = acosf(dotc);
        float so = fmaxf(sinf(omega), 1e-7f);
        const float ts[3] = {0.0f, 1.0f / 3.0f, 2.0f / 3.0f};
        for (int ti = 0; ti < 3; ++ti) {
            float s0 = sinf((1.0f - ts[ti]) * omega);
            float s1 = sinf(ts[ti] * omega);
            float vv[16], vs = 0.f;
            #pragma unroll
            for (int d = 0; d < 16; ++d) { vv[d] = (s0 * hv[d] + s1 * cv[d]) / so; vs += vv[d]*vv[d]; }
            float vi = 1.0f / fmaxf(sqrtf(vs), 1e-12f);
            float* dst = atTab + ((((l * 3 + ti) * 8 + p) * 16 + a) << 4);
            #pragma unroll
            for (int d = 0; d < 16; ++d) dst[d] = vv[d] * vi;
        }
    }
}

// ---------------------------------------------------------------------------
// Prep 2: pack slerp anchors as B-frags: per (l,p): K = d (16 pad 32), N = ti*16+a
// ---------------------------------------------------------------------------
__global__ __launch_bounds__(256) void prep2_kernel(
    const float* __restrict__ atTab, half_t* __restrict__ pkAt)
{
    int i = blockIdx.x * 256 + threadIdx.x;
    if (i >= 24576) return;
    int frag = i >> 9, j = i & 511;
    int lp = frag / 3, nt = frag % 3;
    int l = lp >> 3, p = lp & 7;
    int lane = j >> 4, e = j & 15;
    int k = (e & 7) + ((lane >> 4) << 3) + ((e >> 3) << 4);
    int a = lane & 15;
    float v = (k < 16) ? atTab[((((l * 3 + nt) * 8 + p) * 16 + a) << 4) + k] : 0.0f;
    pkAt[i] = (half_t)v;
}

// ---------------------------------------------------------------------------
// WMMA helpers
// ---------------------------------------------------------------------------
__device__ inline v8f wmma16(v16h a, v16h b, v8f c) {
    return __builtin_amdgcn_wmma_f32_16x16x32_f16(false, a, false, b, (short)0, c, false, false);
}

// 16x32 f16 A-frag from f32 row-major global memory (convert in-register).
// A layout: K(lane,e) = e + (e>=8 ? 8 : 0) + (lane>=16 ? 8 : 0); row = lane&15.
__device__ inline v16h load_a_f32(const float* __restrict__ src, int ld,
                                  int rowBase, int kBase, int lane) {
    const float* p = src + (size_t)(rowBase + (lane & 15)) * ld + kBase + ((lane >> 4) << 3);
    float4 f0 = *(const float4*)(p);
    float4 f1 = *(const float4*)(p + 4);
    float4 f2 = *(const float4*)(p + 16);
    float4 f3 = *(const float4*)(p + 20);
    v16h a;
    a[0]  = (half_t)f0.x; a[1]  = (half_t)f0.y; a[2]  = (half_t)f0.z; a[3]  = (half_t)f0.w;
    a[4]  = (half_t)f1.x; a[5]  = (half_t)f1.y; a[6]  = (half_t)f1.z; a[7]  = (half_t)f1.w;
    a[8]  = (half_t)f2.x; a[9]  = (half_t)f2.y; a[10] = (half_t)f2.z; a[11] = (half_t)f2.w;
    a[12] = (half_t)f3.x; a[13] = (half_t)f3.y; a[14] = (half_t)f3.z; a[15] = (half_t)f3.w;
    return a;
}

// 16x32 f16 A-frag from an f16 LDS tile with row stride `strideH` halfs.
__device__ inline v16h load_a_lds_f16(const half_t* base, int strideH,
                                      int kBase, int lane) {
    const half_t* p = base + (lane & 15) * strideH + kBase + ((lane >> 4) << 3);
    v8h lo  = *(const v8h*)(p);
    v8h hi8 = *(const v8h*)(p + 16);
    v16h a;
    #pragma unroll
    for (int e = 0; e < 8; ++e) { a[e] = lo[e]; a[e + 8] = hi8[e]; }
    return a;
}

// ---------------------------------------------------------------------------
// Fused main kernel: 256 threads = 8 waves; wave w owns rows [blk*128+16w, +16).
// ---------------------------------------------------------------------------
__global__ __launch_bounds__(256) void fused_kernel(
    const float* __restrict__ emb, const float* __restrict__ tri,
    const float* __restrict__ rm,
    const float* __restrict__ embB, const float* __restrict__ triB,
    const float* __restrict__ ctxB,
    const v16h* __restrict__ pkEmb, const v16h* __restrict__ pkTri,
    const v16h* __restrict__ pkCtx, const v16h* __restrict__ pkW1,
    const v16h* __restrict__ pkW2, const v16h* __restrict__ pkAt,
    const float* __restrict__ b1, const float* __restrict__ b2,
    const float* __restrict__ pg, const float* __restrict__ pb,
    const float* __restrict__ gates,
    const float* __restrict__ ng, const float* __restrict__ nb,
    const float* __restrict__ hw1, const float* __restrict__ hb1,
    const float* __restrict__ hw2, const float* __restrict__ hb2,
    const float* __restrict__ sbias,
    float* __restrict__ outMag, float* __restrict__ outMc)
{
    extern __shared__ __align__(16) char dynsm[];
    const int tid = threadIdx.x;
    const int w = tid >> 5, lane = tid & 31;
    const int col = lane & 15, hi = lane >> 4, row = lane & 15;
    const int rowBase = blockIdx.x * 128;        // block rows
    const int wRow = rowBase + w * 16;           // wave rows
    const v8f vzero = {0,0,0,0,0,0,0,0};

    half_t* ctxL  = (half_t*)(dynsm + SM_CTX)  + w * 2048;  // [16][128]
    float*  xT    = (float*)(dynsm + SM_XT)    + w * 2112;  // [16][132]
    float*  lnT   = (float*)(dynsm + SM_LNT)   + w * 2112;  // [16][132]
    half_t* triT  = (half_t*)(dynsm + SM_TRIT) + w * 1152;  // [16][72]
    half_t* hhT   = (half_t*)(dynsm + SM_HHT)  + w * 1152;  // [16][72]
    float*  pwT   = (float*)(dynsm + SM_PWT)   + w * 272;   // [16][17]
    float*  snm   = (float*)(dynsm + SM_SN)    + w * 128;   // [16][8]
    float*  smc   = (float*)(dynsm + SM_MC);                // [128][9] block-wide

    // ======================= stage A: ctx GEMMs ===========================
    v8f accP[4] = {vzero, vzero, vzero, vzero};
    for (int ks = 0; ks < 24; ++ks) {
        v16h a = load_a_f32(emb, DIMV, wRow, ks * 32, lane);
        #pragma unroll
        for (int nt = 0; nt < 4; ++nt)
            accP[nt] = wmma16(a, pkEmb[(ks * 4 + nt) * 32 + lane], accP[nt]);
    }
    v8f accQ[2] = {vzero, vzero};
    for (int ks = 0; ks < 32; ++ks) {
        v16h a = load_a_f32(tri, NAV, wRow, ks * 32, lane);
        #pragma unroll
        for (int nt = 0; nt < 2; ++nt)
            accQ[nt] = wmma16(a, pkTri[(ks * 2 + nt) * 32 + lane], accQ[nt]);
    }
    #pragma unroll
    for (int nt = 0; nt < 4; ++nt) {
        float b = embB[nt * 16 + col];
        #pragma unroll
        for (int v = 0; v < 8; ++v)
            ctxL[(v + 8 * hi) * 128 + nt * 16 + col] = (half_t)(accP[nt][v] + b);
    }
    #pragma unroll
    for (int nt = 0; nt < 2; ++nt) {
        float b = triB[nt * 16 + col];
        #pragma unroll
        for (int v = 0; v < 8; ++v)
            ctxL[(v + 8 * hi) * 128 + 64 + nt * 16 + col] = (half_t)(accQ[nt][v] + b);
    }
    if (lane < 16) ctxL[lane * 128 + 96] = (half_t)rm[wRow + lane];
    for (int idx = lane; idx < 16 * 31; idx += 32)
        ctxL[(idx / 31) * 128 + 97 + idx % 31] = (half_t)0.0f;
    __syncthreads();

    v8f accX[8] = {vzero, vzero, vzero, vzero, vzero, vzero, vzero, vzero};
    #pragma unroll
    for (int ks = 0; ks < 4; ++ks) {
        v16h a = load_a_lds_f16(ctxL, 128, ks * 32, lane);
        #pragma unroll
        for (int nt = 0; nt < 8; ++nt)
            accX[nt] = wmma16(a, pkCtx[(ks * 8 + nt) * 32 + lane], accX[nt]);
    }
    // x tile to LDS (f32, padded stride 132)
    #pragma unroll
    for (int nt = 0; nt < 8; ++nt) {
        float b = ctxB[nt * 16 + col];
        #pragma unroll
        for (int v = 0; v < 8; ++v)
            xT[(v + 8 * hi) * 132 + nt * 16 + col] = accX[nt][v] + b;
    }
    __syncthreads();

    // ======================= relay layers (WMMA) ==========================
    for (int l = 0; l < 2; ++l) {
        // --- LayerNorm over RD=128: 2 lanes per row, shfl-combine halves ---
        const float* xr = xT + row * 132 + hi * 64;
        float part = 0.f;
        for (int i = 0; i < 64; ++i) part += xr[i];
        float mu = (part + __shfl_xor(part, 16)) * (1.0f / 128.0f);
        float vpart = 0.f;
        for (int i = 0; i < 64; ++i) { float d = xr[i] - mu; vpart += d * d; }
        float var = (vpart + __shfl_xor(vpart, 16)) * (1.0f / 128.0f);
        float rstd = rsqrtf(var + 1e-5f);
        float* lr = lnT + row * 132 + hi * 64;
        #pragma unroll
        for (int pi = 0; pi < 4; ++pi) {
            float ss = 0.f;
            #pragma unroll
            for (int d = 0; d < 16; ++d) {
                int c = hi * 64 + pi * 16 + d;
                float v = (xr[pi * 16 + d] - mu) * rstd * ng[l * 128 + c] + nb[l * 128 + c];
                lr[pi * 16 + d] = v;
                ss += v * v;
            }
            snm[row * 8 + hi * 4 + pi] = 1.0f / fmaxf(sqrtf(ss), 1e-12f);
        }
        __syncthreads();

        for (int p = 0; p < KP; ++p) {
            // --- tri = 1 - pn @ at^T : A = pn (K=16 pad 32), B = pkAt ---
            float sn = snm[row * 8 + p];
            const float* lp = lnT + row * 132 + p * 16 + hi * 8;
            float4 f0 = *(const float4*)(lp);
            float4 f1 = *(const float4*)(lp + 4);
            v16h apn;
            apn[0] = (half_t)(f0.x * sn); apn[1] = (half_t)(f0.y * sn);
            apn[2] = (half_t)(f0.z * sn); apn[3] = (half_t)(f0.w * sn);
            apn[4] = (half_t)(f1.x * sn); apn[5] = (half_t)(f1.y * sn);
            apn[6] = (half_t)(f1.z * sn); apn[7] = (half_t)(f1.w * sn);
            #pragma unroll
            for (int e = 8; e < 16; ++e) apn[e] = (half_t)0.0f;
            v8f accT[3] = {vzero, vzero, vzero};
            #pragma unroll
            for (int nt = 0; nt < 3; ++nt)
                accT[nt] = wmma16(apn, pkAt[((l * 8 + p) * 3 + nt) * 32 + lane], accT[nt]);
            #pragma unroll
            for (int nt = 0; nt < 3; ++nt)
                #pragma unroll
                for (int v = 0; v < 8; ++v)
                    triT[(v + 8 * hi) * 72 + nt * 16 + col] = (half_t)(1.0f - accT[nt][v]);
            #pragma unroll
            for (int v = 0; v < 8; ++v)
                triT[(v + 8 * hi) * 72 + 48 + col] = (half_t)0.0f;   // K pad 48..63
            __syncthreads();

            // --- hh = gelu(tri @ w1 + b1): K=64 (2 steps), N=64 (4 tiles) ---
            v8f accH[4] = {vzero, vzero, vzero, vzero};
            #pragma unroll
            for (int ks = 0; ks < 2; ++ks) {
                v16h aT = load_a_lds_f16(triT, 72, ks * 32, lane);
                #pragma unroll
                for (int nt = 0; nt < 4; ++nt)
                    accH[nt] = wmma16(aT, pkW1[(((l * 8 + p) * 2 + ks) * 4 + nt) * 32 + lane], accH[nt]);
            }
            #pragma unroll
            for (int nt = 0; nt < 4; ++nt) {
                float bb = b1[(l * 8 + p) * HV + nt * 16 + col];
                #pragma unroll
                for (int v = 0; v < 8; ++v)
                    hhT[(v + 8 * hi) * 72 + nt * 16 + col] = (half_t)dev_gelu(accH[nt][v] + bb);
            }
            __syncthreads();

            // --- pw_pre = hh @ w2 + b2: K=64 (2 steps), N=16 ---
            v8f accW = vzero;
            #pragma unroll
            for (int ks = 0; ks < 2; ++ks) {
                v16h aH = load_a_lds_f16(hhT, 72, ks * 32, lane);
                accW = wmma16(aH, pkW2[((l * 8 + p) * 2 + ks) * 32 + lane], accW);
            }
            float b2v = b2[(l * 8 + p) * PDV + col];
            #pragma unroll
            for (int v = 0; v < 8; ++v)
                pwT[(v + 8 * hi) * 17 + col] = accW[v] + b2v;
            __syncthreads();

            // --- LN(pw) + gated residual blend (lanes 0..15, one row each) ---
            if (lane < 16) {
                float pmu = 0.f;
                #pragma unroll
                for (int d = 0; d < PDV; ++d) pmu += pwT[lane * 17 + d];
                pmu *= (1.0f / 16.0f);
                float pvar = 0.f;
                #pragma unroll
                for (int d = 0; d < PDV; ++d) { float dd = pwT[lane * 17 + d] - pmu; pvar += dd * dd; }
                pvar *= (1.0f / 16.0f);
                float prs = rsqrtf(pvar + 1e-5f);
                float g = dev_sigmoid(gates[l * 8 + p]);
                #pragma unroll
                for (int d = 0; d < PDV; ++d) {
                    float pwv = (pwT[lane * 17 + d] - pmu) * prs * pg[l * 16 + d] + pb[l * 16 + d];
                    xT[lane * 132 + p * 16 + d] +=
                        g * pwv + (1.0f - g) * lnT[lane * 132 + p * 16 + d];
                }
            }
            __syncthreads();
        }
    }

    // ======================= head: mc per patch ===========================
    #pragma unroll
    for (int kk = 0; kk < 4; ++kk) {
        int k = hi * 4 + kk;
        float mcacc = hb2[k];
        for (int h2 = 0; h2 < 8; ++h2) {
            float acc = hb1[k * 8 + h2];
            #pragma unroll
            for (int d = 0; d < PDV; ++d)
                acc += xT[row * 132 + k * 16 + d] * hw1[(k * 16 + d) * 8 + h2];
            mcacc += dev_gelu(acc) * hw2[k * 8 + h2];
        }
        smc[(w * 16 + row) * 9 + k] = 0.1f + 4.9f * dev_sigmoid(mcacc + sbias[k]);
    }
    __syncthreads();

    // ======================= outputs ======================================
    for (int idx = tid; idx < 128 * 8; idx += 256)
        outMc[(size_t)rowBase * 8 + idx] = smc[(idx >> 3) * 9 + (idx & 7)];
    for (int idx = tid; idx < 128 * 1024; idx += 256) {
        float v = smc[(idx >> 10) * 9 + ((idx & 1023) >> 7)];
        __builtin_nontemporal_store(v, &outMag[(size_t)rowBase * 1024 + idx]);
    }
}

// ---------------------------------------------------------------------------
extern "C" void kernel_launch(void* const* d_in, const int* in_sizes, int n_in,
                              void* d_out, int out_size, void* d_ws, size_t ws_size,
                              hipStream_t stream) {
    (void)in_sizes; (void)n_in; (void)out_size; (void)ws_size;
    const float* emb     = (const float*)d_in[0];
    const float* tri     = (const float*)d_in[1];
    const float* rm      = (const float*)d_in[2];
    const float* embW    = (const float*)d_in[3];
    const float* embB    = (const float*)d_in[4];
    const float* triW    = (const float*)d_in[5];
    const float* triB    = (const float*)d_in[6];
    const float* ctxW    = (const float*)d_in[7];
    const float* ctxB    = (const float*)d_in[8];
    const float* home    = (const float*)d_in[9];
    const float* anchors = (const float*)d_in[10];
    const float* w1      = (const float*)d_in[11];
    const float* b1      = (const float*)d_in[12];
    const float* w2      = (const float*)d_in[13];
    const float* b2      = (const float*)d_in[14];
    const float* pg      = (const float*)d_in[15];
    const float* pb      = (const float*)d_in[16];
    const float* gates   = (const float*)d_in[17];
    const float* ngm     = (const float*)d_in[18];
    const float* nbm     = (const float*)d_in[19];
    const float* hw1     = (const float*)d_in[20];
    const float* hb1     = (const float*)d_in[21];
    const float* hw2     = (const float*)d_in[22];
    const float* hb2     = (const float*)d_in[23];
    const float* sbias   = (const float*)d_in[24];

    char* ws = (char*)d_ws;
    half_t* pkEmb = (half_t*)(ws + WS_PKEMB);
    half_t* pkTri = (half_t*)(ws + WS_PKTRI);
    half_t* pkCtx = (half_t*)(ws + WS_PKCTX);
    half_t* pkW1  = (half_t*)(ws + WS_PKW1);
    half_t* pkW2  = (half_t*)(ws + WS_PKW2);
    float*  atTab = (float*)(ws + WS_ATTAB);
    half_t* pkAt  = (half_t*)(ws + WS_PKAT);

    float* outMag = (float*)d_out;
    float* outMc  = outMag + (size_t)BATCH * NAV;

    prep1_kernel<<<705, 256, 0, stream>>>(embW, triW, ctxW, w1, w2, home, anchors,
                                          pkEmb, pkTri, pkCtx, pkW1, pkW2, atTab);
    prep2_kernel<<<96, 256, 0, stream>>>(atTab, pkAt);

    fused_kernel<<<BATCH / 128, 256, SM_TOTAL, stream>>>(
        emb, tri, rm, embB, triB, ctxB,
        (const v16h*)pkEmb, (const v16h*)pkTri, (const v16h*)pkCtx,
        (const v16h*)pkW1, (const v16h*)pkW2, (const v16h*)pkAt,
        b1, b2, pg, pb, gates, ngm, nbm,
        hw1, hb1, hw2, hb2, sbias, outMag, outMc);
}
